// RSSM_91096256348747
// MI455X (gfx1250) — matrix operand order repaired and checked by reading, not compiled, not run on hardware
//
#include <hip/hip_runtime.h>
#include <hip/hip_bf16.h>

// ---------------------------------------------------------------------------
// RSSM rollout for MI455X (gfx1250): bf16 WMMA GEMMs (f32 accum), weights
// repacked once into WMMA B-fragment order (L2-resident), A tiles staged
// through double-buffered LDS (padded, conflict-free ds_load_b128),
// sequential scan over T=50 steps driven from the host loop.
// All GEMM K dims are padded to multiples of 128 -> single branch-free
// pipelined chunk loop.
// ---------------------------------------------------------------------------

typedef __bf16 bf16_t;
typedef __attribute__((ext_vector_type(16))) __bf16 v16bf;
typedef __attribute__((ext_vector_type(8)))  __bf16 v8bf;
typedef __attribute__((ext_vector_type(8)))  float  v8f;

#define CAT16(lo, hi) __builtin_shufflevector(lo, hi, 0,1,2,3,4,5,6,7,8,9,10,11,12,13,14,15)

#define T_STEPS 50
#define BATCH   512
#define DETER_D 1024
#define STOCH_D 1024
#define ACTOR_D 32
#define NODE_D  1024
#define EMBED_D 1536
#define XK_RAW  1056                 // stoch + actor
#define XK_PAD  1152                 // padded to 9*128 (96 zero cols)

// GEMM geometry
#define GEMM_THREADS 128             // 4 waves
#define BLK_M   32
#define BLK_N   128
#define CHUNK_K 128                  // K staged per LDS round (K % 128 == 0 always)
#define LDS_STRIDE (CHUNK_K + 8)     // +8 bf16 pad -> 272B row stride, no bank conflicts

// ---------------------------------------------------------------------------
// Weight repack: row-major fp32 (Ksrc x N) -> bf16 in WMMA B-fragment tile
// order with K zero-padded to Kp. Tile = 32(K) x 16(N) = 512 elems,
// lane-major. Element e -> k = e + 16*(L>>4), n = 16*ntile + (L&15).
// Tiles ordered ntile-major, ktile-minor -> wave B-frag fetch is 2 coalesced
// b128 loads.
// ---------------------------------------------------------------------------
__global__ __launch_bounds__(256) void rssm_pack_weight(
    const float* __restrict__ W, bf16_t* __restrict__ out,
    int Ksrc, int Kp, int N)
{
    int i = blockIdx.x * 256 + threadIdx.x;
    if (i >= Kp * N) return;
    int ktiles = Kp >> 5;
    int tile   = i >> 9;
    int within = i & 511;
    int L = within >> 4;
    int e = within & 15;
    int ntile = tile / ktiles;
    int ktile = tile - ntile * ktiles;
    int n = ntile * 16 + (L & 15);
    int k = ktile * 32 + e + 16 * (L >> 4);
    out[i] = (k < Ksrc) ? (bf16_t)W[(size_t)k * N + n] : (bf16_t)0.0f;
}

// ---------------------------------------------------------------------------
// bf16 WMMA GEMM: C[M,N] = act(A[M,K] @ B[K,N] + bias), K % 128 == 0.
// A: row-major bf16, staged through double-buffered LDS per 32x128 chunk.
// Bp: fragment-packed bf16, streamed from L2 with coalesced global_load_b128.
// Block = 128 thr = 4 waves; block tile 32(M) x 128(N); wave tile 16 x 64.
// ACT: 0 = none, 1 = ELU.  OUT_BF16: 0 = fp32 out, 1 = bf16 out.
// ---------------------------------------------------------------------------
template <int ACT, int OUT_BF16>
__global__ __launch_bounds__(GEMM_THREADS) void rssm_gemm_wmma(
    const bf16_t* __restrict__ A, const bf16_t* __restrict__ Bp,
    const float* __restrict__ bias, void* __restrict__ Cout,
    int M, int N, int K)
{
    __shared__ bf16_t As[2][BLK_M * LDS_STRIDE];   // 2 x 8704B

    const int tid  = threadIdx.x;
    const int lane = tid & 31;
    const int wave = tid >> 5;
    const int hf   = lane >> 4;     // half-wave select
    const int l15  = lane & 15;

    const int mblk = blockIdx.y * BLK_M;
    const int m0   = mblk + (wave & 1) * 16;
    const int n0   = blockIdx.x * BLK_N + (wave >> 1) * 64;  // wave covers 64 cols
    const int nt0  = n0 >> 4;
    const int ktiles = K >> 5;

    v8f acc0 = {}, acc1 = {}, acc2 = {}, acc3 = {};

    // B fragment pointers (separate named vars -> global_load_b128).
    const bf16_t* bq0 = Bp + ((size_t)(nt0 + 0) * ktiles) * 512 + lane * 16;
    const bf16_t* bq1 = Bp + ((size_t)(nt0 + 1) * ktiles) * 512 + lane * 16;
    const bf16_t* bq2 = Bp + ((size_t)(nt0 + 2) * ktiles) * 512 + lane * 16;
    const bf16_t* bq3 = Bp + ((size_t)(nt0 + 3) * ktiles) * 512 + lane * 16;

    // One k-step: 4 B fragment loads (hoisted, clause-able) + 4 WMMAs.
    auto kstep = [&](v16bf a) {
        v16bf b0 = CAT16(*(const v8bf*)bq0, *(const v8bf*)(bq0 + 8));
        v16bf b1 = CAT16(*(const v8bf*)bq1, *(const v8bf*)(bq1 + 8));
        v16bf b2 = CAT16(*(const v8bf*)bq2, *(const v8bf*)(bq2 + 8));
        v16bf b3 = CAT16(*(const v8bf*)bq3, *(const v8bf*)(bq3 + 8));
        bq0 += 512; bq1 += 512; bq2 += 512; bq3 += 512;
        acc0 = __builtin_amdgcn_wmma_f32_16x16x32_bf16(false, a, false, b0, (short)0, acc0, false, false);
        acc1 = __builtin_amdgcn_wmma_f32_16x16x32_bf16(false, a, false, b1, (short)0, acc1, false, false);
        acc2 = __builtin_amdgcn_wmma_f32_16x16x32_bf16(false, a, false, b2, (short)0, acc2, false, false);
        acc3 = __builtin_amdgcn_wmma_f32_16x16x32_bf16(false, a, false, b3, (short)0, acc3, false, false);
    };

    // Cooperative A staging: 32x128 slab = 512 b128 = exactly 4 per thread.
    v8bf stage0, stage1, stage2, stage3;
    const int srow0 = tid >> 4, scol = (tid & 15) * 8;           // c8 = 16
    auto fetchA = [&](int kc) {
        const bf16_t* base = A + (size_t)(mblk + srow0) * K + kc + scol;
        stage0 = *(const v8bf*)(base);
        stage1 = *(const v8bf*)(base + (size_t)8  * K);          // +8 rows
        stage2 = *(const v8bf*)(base + (size_t)16 * K);
        stage3 = *(const v8bf*)(base + (size_t)24 * K);
    };
    auto commitA = [&](int buf) {
        bf16_t* b = &As[buf][srow0 * LDS_STRIDE + scol];
        *(v8bf*)(b)                   = stage0;
        *(v8bf*)(b +  8 * LDS_STRIDE) = stage1;
        *(v8bf*)(b + 16 * LDS_STRIDE) = stage2;
        *(v8bf*)(b + 24 * LDS_STRIDE) = stage3;
    };

    const int arow = ((wave & 1) * 16 + l15) * LDS_STRIDE + hf * 16;
    auto compute = [&](int buf) {
        const bf16_t* asrc = &As[buf][arow];
        v16bf a0 = CAT16(*(const v8bf*)(asrc +  0), *(const v8bf*)(asrc + 8));
        v16bf a1 = CAT16(*(const v8bf*)(asrc + 32), *(const v8bf*)(asrc + 40));
        v16bf a2 = CAT16(*(const v8bf*)(asrc + 64), *(const v8bf*)(asrc + 72));
        v16bf a3 = CAT16(*(const v8bf*)(asrc + 96), *(const v8bf*)(asrc + 104));
        kstep(a0); kstep(a1); kstep(a2); kstep(a3);
    };

    const int nch = K >> 7;       // # of 128-wide chunks (K % 128 == 0)

    // Prologue: stage chunk 0.
    fetchA(0);
    commitA(0);
    __syncthreads();

    // Steady state: branch-free fetch(next) / compute(cur) / commit(next).
#pragma unroll 1
    for (int c = 0; c < nch - 1; ++c) {
        fetchA((c + 1) << 7);
        compute(c & 1);
        __syncthreads();
        commitA((c + 1) & 1);
        __syncthreads();
    }
    compute((nch - 1) & 1);

#pragma unroll
    for (int j = 0; j < 4; ++j) {
        v8f accv = (j == 0) ? acc0 : (j == 1) ? acc1 : (j == 2) ? acc2 : acc3;
        const int ncol = n0 + j * 16 + l15;
        const float bs = bias[ncol];
#pragma unroll
        for (int r = 0; r < 8; ++r) {
            // C/D layout: lanes 0-15 -> M = r, lanes 16-31 -> M = 8+r; N = l15.
            int row = m0 + r + hf * 8;
            float v = accv[r] + bs;
            if (ACT == 1) v = (v > 0.0f) ? v : (__expf(v) - 1.0f);
            size_t o = (size_t)row * N + ncol;
            if (OUT_BF16) ((bf16_t*)Cout)[o] = (bf16_t)v;
            else          ((float*)Cout)[o]  = v;
        }
    }
}

// ---------------------------------------------------------------------------
// Step-input packing: x = [s*nt | act*nt | 0-pad] -> bf16 [B, 1152];
// hprev = h * nt -> fp32 + bf16 [B,1024].
// ---------------------------------------------------------------------------
__global__ __launch_bounds__(256) void rssm_prestep(
    const bf16_t* __restrict__ s_bf, const float* __restrict__ act_t,
    const float* __restrict__ nt_t, const float* __restrict__ h_f32,
    bf16_t* __restrict__ x_bf, float* __restrict__ hprev_f32,
    bf16_t* __restrict__ hprev_bf)
{
    int i = blockIdx.x * 256 + threadIdx.x;
    int total_x = BATCH * XK_PAD;
    if (i < total_x) {
        int b = i / XK_PAD, c = i - b * XK_PAD;
        float v;
        if (c < STOCH_D)        v = (float)s_bf[(size_t)b * STOCH_D + c] * nt_t[b];
        else if (c < XK_RAW)    v = act_t[(size_t)b * ACTOR_D + (c - STOCH_D)] * nt_t[b];
        else                    v = 0.0f;
        x_bf[i] = (bf16_t)v;
    } else {
        int j = i - total_x;
        if (j < BATCH * DETER_D) {
            int b = j / DETER_D;
            float v = h_f32[j] * nt_t[b];
            hprev_f32[j] = v;
            hprev_bf[j]  = (bf16_t)v;
        }
    }
}

// ---------------------------------------------------------------------------
// GRU gating (torch gate order [r,z,n]): h = (1-z)*n + z*hprev.
// ---------------------------------------------------------------------------
__global__ __launch_bounds__(256) void rssm_gru(
    const float* __restrict__ gi, const float* __restrict__ gh,
    const float* __restrict__ hprev, float* __restrict__ h_f32,
    bf16_t* __restrict__ h_bf, float* __restrict__ deter_out)
{
    int i = blockIdx.x * 256 + threadIdx.x;
    if (i >= BATCH * DETER_D) return;
    int b = i / DETER_D, d = i - b * DETER_D;
    size_t base = (size_t)b * 3 * DETER_D;
    float ir = gi[base + d], iz = gi[base + DETER_D + d], in_ = gi[base + 2 * DETER_D + d];
    float hr = gh[base + d], hz = gh[base + DETER_D + d], hn  = gh[base + 2 * DETER_D + d];
    float r = 1.0f / (1.0f + __expf(-(ir + hr)));
    float z = 1.0f / (1.0f + __expf(-(iz + hz)));
    float n = tanhf(in_ + r * hn);
    float hp = hprev[i];
    float h = (1.0f - z) * n + z * hp;
    h_f32[i] = h;
    h_bf[i] = (bf16_t)h;
    deter_out[i] = h;
}

// ---------------------------------------------------------------------------
// Post-input packing: xp = concat(h, obs) -> bf16 [B, 2560].
// ---------------------------------------------------------------------------
__global__ __launch_bounds__(256) void rssm_pack_xp(
    const bf16_t* __restrict__ h_bf, const float* __restrict__ obs_t,
    bf16_t* __restrict__ xp)
{
    const int XC = DETER_D + EMBED_D;   // 2560
    int i = blockIdx.x * 256 + threadIdx.x;
    if (i >= BATCH * XC) return;
    int b = i / XC, c = i - b * XC;
    xp[i] = (c < DETER_D) ? h_bf[(size_t)b * DETER_D + c]
                          : (bf16_t)obs_t[(size_t)b * EMBED_D + (c - DETER_D)];
}

// ---------------------------------------------------------------------------
// Categorical sample (forward value of the straight-through estimator is the
// one-hot sample): gumbel-perturbed argmax, deterministic counter hash.
// ---------------------------------------------------------------------------
__global__ __launch_bounds__(256) void rssm_sample(
    const float* __restrict__ logits, float* __restrict__ onehot,
    bf16_t* __restrict__ s_bf, unsigned int seed)
{
    int i = blockIdx.x * 256 + threadIdx.x;
    if (i >= BATCH * 32) return;
    int b = i >> 5, cat = i & 31;
    const float* lg = logits + (size_t)b * 1024 + cat * 32;

    unsigned int st = seed * 2654435769u ^ (unsigned int)i * 2246822519u;
    int best = 0; float bestv = -3.0e38f;
#pragma unroll
    for (int c = 0; c < 32; ++c) {
        st = st * 747796405u + 2891336453u;
        unsigned int w = ((st >> ((st >> 28) + 4u)) ^ st) * 277803737u;
        w = (w >> 22) ^ w;
        float u = ((float)w + 0.5f) * (1.0f / 4294967296.0f);
        float g = -__logf(-__logf(u));
        float v = lg[c] + g;
        if (v > bestv) { bestv = v; best = c; }
    }
    float* out = onehot + (size_t)b * 1024 + cat * 32;
#pragma unroll
    for (int c = 0; c < 32; ++c) {
        float o = (c == best) ? 1.0f : 0.0f;
        out[c] = o;
        if (s_bf) s_bf[(size_t)b * 1024 + cat * 32 + c] = (bf16_t)o;
    }
}

// ---------------------------------------------------------------------------
// Host driver
// ---------------------------------------------------------------------------
extern "C" void kernel_launch(void* const* d_in, const int* in_sizes, int n_in,
                              void* d_out, int out_size, void* d_ws, size_t ws_size,
                              hipStream_t stream)
{
    (void)in_sizes; (void)n_in; (void)out_size; (void)ws_size;

    const float* obs   = (const float*)d_in[0];   // (T,B,1536)
    const float* act   = (const float*)d_in[1];   // (T,B,32)
    const float* nt    = (const float*)d_in[2];   // (T,B,1)
    const float* emb_W = (const float*)d_in[3];   // (1056,1024)
    const float* emb_b = (const float*)d_in[4];
    const float* w_ih  = (const float*)d_in[5];   // (1024,3072)
    const float* w_hh  = (const float*)d_in[6];
    const float* b_ih  = (const float*)d_in[7];
    const float* b_hh  = (const float*)d_in[8];
    const float* pr_W1 = (const float*)d_in[9];   // (1024,1024)
    const float* pr_b1 = (const float*)d_in[10];
    const float* pr_W2 = (const float*)d_in[11];  // (1024,1024)
    const float* pr_b2 = (const float*)d_in[12];
    const float* po_W1 = (const float*)d_in[13];  // (2560,1024)
    const float* po_b1 = (const float*)d_in[14];
    const float* po_W2 = (const float*)d_in[15];  // (1024,1024)
    const float* po_b2 = (const float*)d_in[16];

    char* ws = (char*)d_ws;
    size_t off = 0;
    auto alloc = [&](size_t bytes) { char* p = ws + off; off += (bytes + 255) & ~(size_t)255; return p; };

    bf16_t* WP_EMB = (bf16_t*)alloc((size_t)XK_PAD * 1024 * 2);
    bf16_t* WP_IH  = (bf16_t*)alloc((size_t)1024 * 3072 * 2);
    bf16_t* WP_HH  = (bf16_t*)alloc((size_t)1024 * 3072 * 2);
    bf16_t* WP_PR1 = (bf16_t*)alloc((size_t)1024 * 1024 * 2);
    bf16_t* WP_PR2 = (bf16_t*)alloc((size_t)1024 * 1024 * 2);
    bf16_t* WP_PO1 = (bf16_t*)alloc((size_t)2560 * 1024 * 2);
    bf16_t* WP_PO2 = (bf16_t*)alloc((size_t)1024 * 1024 * 2);
    bf16_t* X_BF   = (bf16_t*)alloc((size_t)BATCH * XK_PAD * 2);
    bf16_t* XP_BF  = (bf16_t*)alloc((size_t)BATCH * 2560 * 2);
    bf16_t* E_BF   = (bf16_t*)alloc((size_t)BATCH * 1024 * 2);
    bf16_t* P1_BF  = (bf16_t*)alloc((size_t)BATCH * 1024 * 2);
    bf16_t* Q1_BF  = (bf16_t*)alloc((size_t)BATCH * 1024 * 2);
    bf16_t* H_BF   = (bf16_t*)alloc((size_t)BATCH * 1024 * 2);
    bf16_t* HP_BF  = (bf16_t*)alloc((size_t)BATCH * 1024 * 2);
    bf16_t* S_BF   = (bf16_t*)alloc((size_t)BATCH * 1024 * 2);
    float*  H_F32  = (float*) alloc((size_t)BATCH * 1024 * 4);
    float*  HP_F32 = (float*) alloc((size_t)BATCH * 1024 * 4);
    float*  GI     = (float*) alloc((size_t)BATCH * 3072 * 4);
    float*  GH     = (float*) alloc((size_t)BATCH * 3072 * 4);

    // One-shot (per launch) weight repack into WMMA fragment order.
    auto packW = [&](const float* W, bf16_t* dst, int Ksrc, int Kp, int N) {
        rssm_pack_weight<<<(Kp * N + 255) / 256, 256, 0, stream>>>(W, dst, Ksrc, Kp, N);
    };
    packW(emb_W, WP_EMB, XK_RAW, XK_PAD, 1024);
    packW(w_ih,  WP_IH,  1024, 1024, 3072);
    packW(w_hh,  WP_HH,  1024, 1024, 3072);
    packW(pr_W1, WP_PR1, 1024, 1024, 1024);
    packW(pr_W2, WP_PR2, 1024, 1024, 1024);
    packW(po_W1, WP_PO1, 2560, 2560, 1024);
    packW(po_W2, WP_PO2, 1024, 1024, 1024);

    // Initial carry: h0 = 0, s0 = 0.
    hipMemsetAsync(H_F32, 0, (size_t)BATCH * 1024 * 4, stream);
    hipMemsetAsync(S_BF,  0, (size_t)BATCH * 1024 * 2, stream);

    float* out = (float*)d_out;
    const size_t OSZ = (size_t)T_STEPS * BATCH * 1024;

    const dim3 g1024(1024 / BLK_N, BATCH / BLK_M);   // N=1024 GEMMs: (8,16)
    const dim3 g3072(3072 / BLK_N, BATCH / BLK_M);   // N=3072 GEMMs: (24,16)
    const int  PRE_BLOCKS  = (BATCH * (XK_PAD + 1024) + 255) / 256;
    const int  PW_BLOCKS   = (BATCH * 1024 + 255) / 256;
    const int  XP_BLOCKS   = (BATCH * 2560 + 255) / 256;
    const int  SAMP_BLOCKS = (BATCH * 32 + 255) / 256;

    for (int t = 0; t < T_STEPS; ++t) {
        const float* obs_t = obs + (size_t)t * BATCH * EMBED_D;
        const float* act_t = act + (size_t)t * BATCH * ACTOR_D;
        const float* nt_t  = nt  + (size_t)t * BATCH;
        float* deter_t = out + 0 * OSZ + (size_t)t * BATCH * 1024;
        float* prst_t  = out + 1 * OSZ + (size_t)t * BATCH * 1024;
        float* prlg_t  = out + 2 * OSZ + (size_t)t * BATCH * 1024;
        float* post_t  = out + 3 * OSZ + (size_t)t * BATCH * 1024;
        float* polg_t  = out + 4 * OSZ + (size_t)t * BATCH * 1024;

        // x = [s*nt | a*nt | 0]; hprev = h*nt
        rssm_prestep<<<PRE_BLOCKS, 256, 0, stream>>>(S_BF, act_t, nt_t, H_F32,
                                                     X_BF, HP_F32, HP_BF);
        // e = elu(x @ emb_W + emb_b)   (K padded to 1152)
        rssm_gemm_wmma<1, 1><<<g1024, GEMM_THREADS, 0, stream>>>(
            X_BF, WP_EMB, emb_b, E_BF, BATCH, 1024, XK_PAD);
        // gi = e @ w_ih + b_ih ; gh = hprev @ w_hh + b_hh
        rssm_gemm_wmma<0, 0><<<g3072, GEMM_THREADS, 0, stream>>>(
            E_BF, WP_IH, b_ih, GI, BATCH, 3072, 1024);
        rssm_gemm_wmma<0, 0><<<g3072, GEMM_THREADS, 0, stream>>>(
            HP_BF, WP_HH, b_hh, GH, BATCH, 3072, 1024);
        // GRU gating -> h
        rssm_gru<<<PW_BLOCKS, 256, 0, stream>>>(GI, GH, HP_F32, H_F32, H_BF, deter_t);
        // prior chain
        rssm_gemm_wmma<1, 1><<<g1024, GEMM_THREADS, 0, stream>>>(
            H_BF, WP_PR1, pr_b1, P1_BF, BATCH, 1024, 1024);
        rssm_gemm_wmma<0, 0><<<g1024, GEMM_THREADS, 0, stream>>>(
            P1_BF, WP_PR2, pr_b2, prlg_t, BATCH, 1024, 1024);
        rssm_sample<<<SAMP_BLOCKS, 256, 0, stream>>>(prlg_t, prst_t, (bf16_t*)nullptr,
                                                     (unsigned int)(2 * t));
        // posterior chain
        rssm_pack_xp<<<XP_BLOCKS, 256, 0, stream>>>(H_BF, obs_t, XP_BF);
        rssm_gemm_wmma<1, 1><<<g1024, GEMM_THREADS, 0, stream>>>(
            XP_BF, WP_PO1, po_b1, Q1_BF, BATCH, 1024, 2560);
        rssm_gemm_wmma<0, 0><<<g1024, GEMM_THREADS, 0, stream>>>(
            Q1_BF, WP_PO2, po_b2, polg_t, BATCH, 1024, 1024);
        rssm_sample<<<SAMP_BLOCKS, 256, 0, stream>>>(polg_t, post_t, S_BF,
                                                     (unsigned int)(2 * t + 1));
    }
}
